// ResidualBlock_34677565948740
// MI455X (gfx1250) — matrix operand order repaired and checked by reading, not compiled, not run on hardware
//
#include <hip/hip_runtime.h>

#define NN 16000
#define EE 256000
#define CCH 64
#define KF 125
#define CAP (EE * 8 + KF * 16)   // padded bucket capacity = 2,050,000
#define NTILES (CAP / 16)        // 128,125 tiles max

typedef __attribute__((ext_vector_type(16))) __bf16 v16bf;
typedef __attribute__((ext_vector_type(8)))  __bf16 v8bf;
typedef __attribute__((ext_vector_type(8)))  float  v8f;

static __device__ __forceinline__ void faa(float* p, float v) { unsafeAtomicAdd(p, v); }

// ---------------- conversion kernels ----------------
__global__ void k_cvt_bf16(const float* __restrict__ s, __bf16* __restrict__ d, int n) {
  int i = blockIdx.x * blockDim.x + threadIdx.x;
  if (i < n) d[i] = (__bf16)s[i];
}

// wt[k][o][i] = W[k][i][o]  (bf16 transpose so B-operand columns are contiguous)
__global__ void k_cvt_wT(const float* __restrict__ W, __bf16* __restrict__ wt, int K) {
  int i = blockIdx.x * blockDim.x + threadIdx.x;
  if (i >= K * 4096) return;
  int k = i >> 12, rem = i & 4095, o = rem >> 6, ii = rem & 63;
  wt[i] = (__bf16)W[(k << 12) + (ii << 6) + o];
}

// ---------------- spline basis bucketing ----------------
static __device__ __forceinline__ void basis_setup(const float* __restrict__ attr, int e,
                                                   float* fr, int* lo) {
#pragma unroll
  for (int d = 0; d < 3; ++d) {
    float p = attr[e * 3 + d] * 4.0f;
    float fl = floorf(p);
    lo[d] = (int)fl;
    fr[d] = p - fl;
  }
}

__global__ void k_basis_count(const int* __restrict__ ei, const float* __restrict__ attr,
                              int* __restrict__ counts, float* __restrict__ deg) {
  int e = blockIdx.x * blockDim.x + threadIdx.x;
  if (e >= EE) return;
  faa(&deg[ei[EE + e]], 1.0f);
  float fr[3]; int lo[3];
  basis_setup(attr, e, fr, lo);
#pragma unroll
  for (int s = 0; s < 8; ++s) {
    int kidx = 0, mul = 1;
#pragma unroll
    for (int d = 0; d < 3; ++d) {
      int bit = (s >> d) & 1;
      int q = lo[d] + bit; q = q < 0 ? 0 : (q > 4 ? 4 : q);
      kidx += q * mul; mul *= 5;
    }
    atomicAdd(&counts[kidx], 1);
  }
}

__global__ void k_scan(const int* __restrict__ counts, int* __restrict__ offs,
                       int* __restrict__ cursor) {
  if (blockIdx.x == 0 && threadIdx.x == 0) {
    int run = 0;
    for (int k = 0; k < KF; ++k) {
      offs[k] = run; cursor[k] = run;
      run += (counts[k] + 15) & ~15;   // pad each bucket to a 16-row tile
    }
    offs[KF] = run;
  }
}

__global__ void k_basis_scatter(const int* __restrict__ ei, const float* __restrict__ attr,
                                int* __restrict__ cursor, int* __restrict__ srcA,
                                int* __restrict__ tgtA, float* __restrict__ bwA) {
  int e = blockIdx.x * blockDim.x + threadIdx.x;
  if (e >= EE) return;
  int src = ei[e], tgt = ei[EE + e];
  float fr[3]; int lo[3];
  basis_setup(attr, e, fr, lo);
#pragma unroll
  for (int s = 0; s < 8; ++s) {
    int kidx = 0, mul = 1; float b = 1.0f;
#pragma unroll
    for (int d = 0; d < 3; ++d) {
      int bit = (s >> d) & 1;
      int q = lo[d] + bit; q = q < 0 ? 0 : (q > 4 ? 4 : q);
      kidx += q * mul; mul *= 5;
      b *= bit ? fr[d] : 1.0f - fr[d];
    }
    int pos = atomicAdd(&cursor[kidx], 1);
    srcA[pos] = src; tgtA[pos] = tgt; bwA[pos] = b;
  }
}

// ---------------- WMMA 16x64 tile: D[16,64] = A[16,64] @ W[64,64] ----------------
// A-operand (16-bit, 16x32): lanes 0-15 row M=lane, K {0..15}+ks*32;
// lanes 16-31 same row, K {16..31}+ks*32.
// B-operand: lane column N = nt*16+(lane&15); 16 contiguous K at (lane>>4)*16+ks*32.
static __device__ __forceinline__ void mm16x64(const __bf16* __restrict__ arow,
                                               const __bf16* __restrict__ w,
                                               int lh, int hb, v8f* c) {
#pragma unroll
  for (int ks = 0; ks < 2; ++ks) {
    v8bf alo = *(const v8bf*)(arow + ks * 32 + hb * 16);
    v8bf ahi = *(const v8bf*)(arow + ks * 32 + hb * 16 + 8);
    v16bf a = __builtin_shufflevector(alo, ahi, 0, 1, 2, 3, 4, 5, 6, 7,
                                      8, 9, 10, 11, 12, 13, 14, 15);
    int kb = ks * 32 + hb * 16;
#pragma unroll
    for (int nt = 0; nt < 4; ++nt) {
      v16bf b = *(const v16bf*)(w + (nt * 16 + lh) * 64 + kb);
      c[nt] = __builtin_amdgcn_wmma_f32_16x16x32_bf16(false, a, false, b,
                                                      (short)0, c[nt], false, false);
    }
  }
}

// one wave per 16-row bucket tile; scaled atomic scatter of the result rows
__global__ void k_bucket_gemm(const __bf16* __restrict__ xb, const __bf16* __restrict__ wt,
                              const int* __restrict__ offs, const int* __restrict__ srcA,
                              const int* __restrict__ tgtA, const float* __restrict__ bwA,
                              float* __restrict__ acc) {
  int t0 = blockIdx.x * 16;
  if (t0 >= offs[KF]) return;            // wave-uniform: EXEC all-1 at WMMA
  int lo = 0, hi = KF;
  while (hi - lo > 1) { int mid = (lo + hi) >> 1; if (offs[mid] <= t0) lo = mid; else hi = mid; }
  const __bf16* w = wt + (lo << 12);
  int lane = threadIdx.x, lh = lane & 15, hb = lane >> 4;
  int srow = srcA[t0 + lh];
  if (bwA[t0 + lh] == 0.0f) srow = 0;    // padding rows: stale src guarded out
  v8f c[4];
#pragma unroll
  for (int nt = 0; nt < 4; ++nt)
#pragma unroll
    for (int j = 0; j < 8; ++j) c[nt][j] = 0.0f;
  mm16x64(xb + srow * 64, w, lh, hb, c);
#pragma unroll
  for (int r = 0; r < 8; ++r) {
    int M = t0 + r + hb * 8;
    float bw = bwA[M];
    if (bw != 0.0f) {
      float* o = acc + tgtA[M] * 64 + lh;
      faa(o,      c[0][r] * bw);
      faa(o + 16, c[1][r] * bw);
      faa(o + 32, c[2][r] * bw);
      faa(o + 48, c[3][r] * bw);
    }
  }
}

// dense row-tile GEMM (root weights / shortcut); addMode: 0=store 1=accumulate
__global__ void k_dense_gemm(const __bf16* __restrict__ xb, const __bf16* __restrict__ wt,
                             float* __restrict__ out, int addMode) {
  int base = blockIdx.x * 16;
  int lane = threadIdx.x, lh = lane & 15, hb = lane >> 4;
  v8f c[4];
#pragma unroll
  for (int nt = 0; nt < 4; ++nt)
#pragma unroll
    for (int j = 0; j < 8; ++j) c[nt][j] = 0.0f;
  mm16x64(xb + (base + lh) * 64, wt, lh, hb, c);
#pragma unroll
  for (int r = 0; r < 8; ++r) {
    float* o = out + (base + r + hb * 8) * 64 + lh;
    if (addMode) { o[0] += c[0][r]; o[16] += c[1][r]; o[32] += c[2][r]; o[48] += c[3][r]; }
    else         { o[0]  = c[0][r]; o[16]  = c[1][r]; o[32]  = c[2][r]; o[48]  = c[3][r]; }
  }
}

// ---------------- shortcut aggregation ----------------
__global__ void k_agg_scatter(const int* __restrict__ ei, const float* __restrict__ x,
                              float* __restrict__ aggacc) {
  int i = blockIdx.x * blockDim.x + threadIdx.x;
  if (i >= EE * 64) return;
  int e = i >> 6, c = i & 63;
  faa(&aggacc[ei[EE + e] * 64 + c], x[ei[e] * 64 + c]);
}

__global__ void k_agg_cvt(const float* __restrict__ aggacc, const float* __restrict__ deg,
                          __bf16* __restrict__ aggbf) {
  int i = blockIdx.x * blockDim.x + threadIdx.x;
  if (i >= NN * 64) return;
  aggbf[i] = (__bf16)(aggacc[i] / fmaxf(deg[i >> 6], 1.0f));
}

// ---------------- epilogues: bias + deg-mean + BN stats, BN apply ----------------
__global__ void k_finalize(float* __restrict__ pre, const float* __restrict__ acc,
                           const float* __restrict__ deg, const float* __restrict__ bias,
                           float* __restrict__ s1, float* __restrict__ s2, int useAcc) {
  __shared__ float sh1[256], sh2[256];
  int t = threadIdx.x, c = t & 63;
  float ls = 0.f, lq = 0.f;
  for (int idx = blockIdx.x * 256 + t; idx < NN * 64; idx += gridDim.x * 256) {
    float v = pre[idx] + bias[c];
    if (useAcc) v += acc[idx] / fmaxf(deg[idx >> 6], 1.0f);
    pre[idx] = v; ls += v; lq += v * v;
  }
  sh1[t] = ls; sh2[t] = lq;
  __syncthreads();
  if (t < 64) {
    faa(&s1[t], sh1[t] + sh1[t + 64] + sh1[t + 128] + sh1[t + 192]);
    faa(&s2[t], sh2[t] + sh2[t + 64] + sh2[t + 128] + sh2[t + 192]);
  }
}

static __device__ __forceinline__ float bn_apply(float v, float s1, float s2,
                                                 float g, float be) {
  float mu = s1 * (1.0f / NN);
  float var = s2 * (1.0f / NN) - mu * mu;
  return g * (v - mu) * rsqrtf(var + 1e-5f) + be;
}

__global__ void k_bn_elu_bf16(const float* __restrict__ pre, const float* __restrict__ s1,
                              const float* __restrict__ s2, const float* __restrict__ g,
                              const float* __restrict__ be, __bf16* __restrict__ hb) {
  int i = blockIdx.x * blockDim.x + threadIdx.x;
  if (i >= NN * 64) return;
  int c = i & 63;
  float xn = bn_apply(pre[i], s1[c], s2[c], g[c], be[c]);
  hb[i] = (__bf16)(xn > 0.f ? xn : expm1f(xn));
}

__global__ void k_final(const float* __restrict__ pre2, const float* __restrict__ sp,
                        const float* __restrict__ s1a, const float* __restrict__ s2a,
                        const float* __restrict__ g2, const float* __restrict__ be2,
                        const float* __restrict__ s1b, const float* __restrict__ s2b,
                        const float* __restrict__ gs, const float* __restrict__ bes,
                        float* __restrict__ out) {
  int i = blockIdx.x * blockDim.x + threadIdx.x;
  if (i >= NN * 64) return;
  int c = i & 63;
  float v = bn_apply(pre2[i], s1a[c], s2a[c], g2[c], be2[c]) +
            bn_apply(sp[i],   s1b[c], s2b[c], gs[c], bes[c]);
  out[i] = v > 0.f ? v : expm1f(v);
}

// ---------------- host ----------------
extern "C" void kernel_launch(void* const* d_in, const int* in_sizes, int n_in,
                              void* d_out, int out_size, void* d_ws, size_t ws_size,
                              hipStream_t stream) {
  (void)in_sizes; (void)n_in; (void)out_size; (void)ws_size;
  const float* x     = (const float*)d_in[0];
  const int*   ei    = (const int*)d_in[1];
  const float* attr  = (const float*)d_in[2];
  const float* W1    = (const float*)d_in[3];
  const float* root1 = (const float*)d_in[4];
  const float* b1    = (const float*)d_in[5];
  const float* g1    = (const float*)d_in[6];
  const float* be1   = (const float*)d_in[7];
  const float* W2    = (const float*)d_in[8];
  const float* root2 = (const float*)d_in[9];
  const float* b2    = (const float*)d_in[10];
  const float* g2    = (const float*)d_in[11];
  const float* be2   = (const float*)d_in[12];
  const float* Ws    = (const float*)d_in[13];
  const float* roots = (const float*)d_in[14];
  const float* bs    = (const float*)d_in[15];
  const float* gs    = (const float*)d_in[16];
  const float* bes   = (const float*)d_in[17];
  float* out = (float*)d_out;

  char* p = (char*)d_ws;
  auto alloc = [&](size_t bytes) -> void* {
    void* r = (void*)p; p += (bytes + 255) & ~(size_t)255; return r;
  };
  int*    counts = (int*)alloc(128 * 4);
  int*    offs   = (int*)alloc(128 * 4);
  int*    cursor = (int*)alloc(128 * 4);
  float*  stats  = (float*)alloc(6 * 64 * 4);
  float*  deg    = (float*)alloc(NN * 4);
  int*    srcA   = (int*)alloc((size_t)CAP * 4);
  int*    tgtA   = (int*)alloc((size_t)CAP * 4);
  float*  bwA    = (float*)alloc((size_t)CAP * 4);
  __bf16* xbf    = (__bf16*)alloc((size_t)NN * 64 * 2);
  __bf16* hbf    = (__bf16*)alloc((size_t)NN * 64 * 2);
  __bf16* aggbf  = (__bf16*)alloc((size_t)NN * 64 * 2);
  __bf16* w1t    = (__bf16*)alloc((size_t)KF * 4096 * 2);
  __bf16* w2t    = (__bf16*)alloc((size_t)KF * 4096 * 2);
  __bf16* wst    = (__bf16*)alloc(4096 * 2);
  __bf16* r1t    = (__bf16*)alloc(4096 * 2);
  __bf16* r2t    = (__bf16*)alloc(4096 * 2);
  __bf16* rst    = (__bf16*)alloc(4096 * 2);
  float*  acc    = (float*)alloc((size_t)NN * 64 * 4);
  float*  pre1   = (float*)alloc((size_t)NN * 64 * 4);
  float*  pre2   = (float*)alloc((size_t)NN * 64 * 4);
  float*  scpre  = (float*)alloc((size_t)NN * 64 * 4);
  float*  aggacc = (float*)alloc((size_t)NN * 64 * 4);

  hipMemsetAsync(counts, 0, 128 * 4, stream);
  hipMemsetAsync(stats, 0, 6 * 64 * 4, stream);
  hipMemsetAsync(deg, 0, NN * 4, stream);
  hipMemsetAsync(bwA, 0, (size_t)CAP * 4, stream);
  hipMemsetAsync(acc, 0, (size_t)NN * 64 * 4, stream);
  hipMemsetAsync(aggacc, 0, (size_t)NN * 64 * 4, stream);

  const int NC = NN * 64;
  k_cvt_bf16<<<(NC + 255) / 256, 256, 0, stream>>>(x, xbf, NC);
  k_cvt_wT<<<(KF * 4096 + 255) / 256, 256, 0, stream>>>(W1, w1t, KF);
  k_cvt_wT<<<(KF * 4096 + 255) / 256, 256, 0, stream>>>(W2, w2t, KF);
  k_cvt_wT<<<16, 256, 0, stream>>>(Ws, wst, 1);
  k_cvt_wT<<<16, 256, 0, stream>>>(root1, r1t, 1);
  k_cvt_wT<<<16, 256, 0, stream>>>(root2, r2t, 1);
  k_cvt_wT<<<16, 256, 0, stream>>>(roots, rst, 1);

  k_basis_count<<<(EE + 255) / 256, 256, 0, stream>>>(ei, attr, counts, deg);
  k_scan<<<1, 1, 0, stream>>>(counts, offs, cursor);
  k_basis_scatter<<<(EE + 255) / 256, 256, 0, stream>>>(ei, attr, cursor, srcA, tgtA, bwA);
  k_agg_scatter<<<(EE * 64 + 255) / 256, 256, 0, stream>>>(ei, x, aggacc);

  // conv1
  k_bucket_gemm<<<NTILES, 32, 0, stream>>>(xbf, w1t, offs, srcA, tgtA, bwA, acc);
  k_dense_gemm<<<NN / 16, 32, 0, stream>>>(xbf, r1t, pre1, 0);
  k_finalize<<<256, 256, 0, stream>>>(pre1, acc, deg, b1, stats + 0, stats + 64, 1);
  k_bn_elu_bf16<<<(NC + 255) / 256, 256, 0, stream>>>(pre1, stats + 0, stats + 64, g1, be1, hbf);

  // conv2
  hipMemsetAsync(acc, 0, (size_t)NN * 64 * 4, stream);
  k_bucket_gemm<<<NTILES, 32, 0, stream>>>(hbf, w2t, offs, srcA, tgtA, bwA, acc);
  k_dense_gemm<<<NN / 16, 32, 0, stream>>>(hbf, r2t, pre2, 0);
  k_finalize<<<256, 256, 0, stream>>>(pre2, acc, deg, b2, stats + 128, stats + 192, 1);

  // shortcut (ks=1): mean-aggregate then GEMMs
  k_agg_cvt<<<(NC + 255) / 256, 256, 0, stream>>>(aggacc, deg, aggbf);
  k_dense_gemm<<<NN / 16, 32, 0, stream>>>(aggbf, wst, scpre, 0);
  k_dense_gemm<<<NN / 16, 32, 0, stream>>>(xbf, rst, scpre, 1);
  k_finalize<<<256, 256, 0, stream>>>(scpre, nullptr, nullptr, bs, stats + 256, stats + 320, 0);

  k_final<<<(NC + 255) / 256, 256, 0, stream>>>(pre2, scpre, stats + 128, stats + 192, g2, be2,
                                                stats + 256, stats + 320, gs, bes, out);
}
